// Model_79955111182621
// MI455X (gfx1250) — compile-verified, matrix operand order checked by reading, not run on hardware
//
#include <hip/hip_runtime.h>

typedef __attribute__((ext_vector_type(16))) _Float16 v16h;
typedef __attribute__((ext_vector_type(8)))  _Float16 v8h;
typedef __attribute__((ext_vector_type(8)))  float    v8f;
typedef __attribute__((ext_vector_type(4)))  unsigned u32x4;
typedef __attribute__((ext_vector_type(8)))  unsigned u32x8;

#define F_INRELU  1
#define F_OUTRELU 2
#define F_TRANS   4

// ---------------------------------------------------------------- f32 -> f16
__global__ __launch_bounds__(256) void f32_to_f16_k(const float* __restrict__ s,
                                                    _Float16* __restrict__ d, int n) {
  int i = blockIdx.x * blockDim.x + threadIdx.x;
  int stride = gridDim.x * blockDim.x;
  for (; i < n; i += stride) d[i] = (_Float16)s[i];
}

// ------------------------------------------------- implicit-GEMM conv (WMMA)
// Each wave computes a 16(M)x64(N) strip: one A im2col fragment feeds 4 WMMAs.
// The 64x32 f16 weight chunk is staged into LDS once per block:
//  - normal conv: one TDM tensor_load_to_lds (2D tile, OOB zero-fills tails)
//  - transposed:  per-dword async global->LDS copies (K walk crosses ci rows)
__global__ __launch_bounds__(128) void conv_wmma_k(
    const float* __restrict__ in, const _Float16* __restrict__ wf,
    const float* __restrict__ bias, const float* __restrict__ resid,
    float* __restrict__ out,
    int NB, int Cin, int IH, int IW, int Cout, int OH, int OW,
    int KH, int KW, int SH, int SW, int PH, int PW, int flags)
{
  __shared__ _Float16 Btile[64 * 32];          // 4KB staging tile (n x kk)

  const int lane   = threadIdx.x & 31;
  const int wave   = threadIdx.x >> 5;
  const int Mtot   = NB * OH * OW;
  const int tilesM = (Mtot + 15) >> 4;
  int tileM = blockIdx.x * (blockDim.x >> 5) + wave;
  if (tileM >= tilesM) tileM = tilesM - 1;     // keep wave alive for barriers
  const int nbase  = blockIdx.y * 64;
  const int lane16 = lane & 15, half = lane >> 4;
  const int KHW  = KH * KW;
  const int Ktot = Cin * KHW;                  // even for every layer here

  const int  mrow = tileM * 16 + lane16;
  const bool mOK  = mrow < Mtot;
  int ab = 0, aoh = 0, aow = 0;
  {
    const int ohw = OH * OW;
    const int mm = mOK ? mrow : 0;
    ab = mm / ohw; int r = mm - ab * ohw; aoh = r / OW; aow = r - aoh * OW;
  }
  const bool trans  = (flags & F_TRANS) != 0;
  const bool inRelu = (flags & F_INRELU) != 0;

  v8f acc[4];
#pragma unroll
  for (int nt = 0; nt < 4; ++nt)
    acc[nt] = (v8f){0.f, 0.f, 0.f, 0.f, 0.f, 0.f, 0.f, 0.f};

  for (int k0 = 0; k0 < Ktot; k0 += 32) {
    if (!trans) {
      // ---- TDM: one 2D tile load (64 rows x 32 halves), zero-fill OOB
      if (wave == 0) {
        const _Float16* gp = wf + (size_t)nbase * Ktot + k0;
        const unsigned long long ga = (unsigned long long)(uintptr_t)gp;
        const unsigned ldsA = (unsigned)(uintptr_t)(void*)&Btile[0];
        const unsigned td0 = (unsigned)(Ktot - k0);     // remaining K extent
        const unsigned td1 = (unsigned)(Cout - nbase);  // remaining N extent
        u32x4 g0;
        g0[0] = 1u;                                     // count=1, user mode
        g0[1] = ldsA;                                   // lds_addr
        g0[2] = (unsigned)ga;                           // global_addr[31:0]
        g0[3] = ((unsigned)(ga >> 32) & 0x01FFFFFFu) | 0x80000000u; // +type=2
        u32x8 g1;
        g1[0] = 0x00010000u;                            // data_size=2 bytes
        g1[1] = (td0 & 0xFFFFu) << 16;                  // tensor_dim0 lo
        g1[2] = ((td0 >> 16) & 0xFFFFu) | ((td1 & 0xFFFFu) << 16);
        g1[3] = ((td1 >> 16) & 0xFFFFu) | (32u << 16);  // tile_dim0 = 32
        g1[4] = 64u;                                    // tile_dim1 = 64
        g1[5] = (unsigned)Ktot;                         // dim0 stride (elems)
        g1[6] = 0u;
        g1[7] = 0u;
        asm volatile("tensor_load_to_lds %0, %1" :: "s"(g0), "s"(g1) : "memory");
        __builtin_amdgcn_s_wait_tensorcnt(0);
      }
    } else {
      // ---- async global->LDS per dword (64 rows x 16 dwords = 1024 dwords)
      for (int d = threadIdx.x; d < 1024; d += 128) {
        const int nl = d >> 4;                 // local n row 0..63
        const int j  = d & 15;                 // dword within row
        const int k  = k0 + j * 2;
        const int ng = nbase + nl;
        if (ng < Cout && k < Ktot) {
          const int ci = k / KHW; const int rr = k - ci * KHW;
          const _Float16* gp = wf + ((size_t)ci * Cout + ng) * KHW + rr;
          const unsigned ldsAddr = (unsigned)(uintptr_t)(void*)&Btile[nl * 32 + j * 2];
          const unsigned long long ga = (unsigned long long)(uintptr_t)gp;
          asm volatile("global_load_async_to_lds_b32 %0, %1, off"
                       :: "v"(ldsAddr), "v"(ga) : "memory");
        } else {
          *(unsigned*)&Btile[nl * 32 + j * 2] = 0u;
        }
      }
      asm volatile("s_wait_asynccnt 0x0" ::: "memory");
    }
    __syncthreads();

    // ---- A fragment: per-lane im2col gather (f32 -> f16), shared by 4 WMMAs
    v16h A = {};
#pragma unroll
    for (int v = 0; v < 8; ++v) {
#pragma unroll
      for (int e = 0; e < 2; ++e) {
        const int kk = ((v & 4) << 2) + half * 8 + (v & 3) * 2 + e;
        const int k  = k0 + kk;
        float av = 0.f;
        if (mOK && k < Ktot) {
          const int ci = k / KHW; const int rr = k - ci * KHW;
          const int kh = rr / KW; const int kw = rr - kh * KW;
          int ih, iw; bool ok;
          if (trans) {              // ConvTranspose gather: oh = ih*S - P + kh
            const int t = aoh + PH - kh, u = aow + PW - kw;
            ih = t / SH; iw = u / SW;
            ok = (t >= 0) && (u >= 0) && (t == ih * SH) && (u == iw * SW)
                 && (ih < IH) && (iw < IW);
          } else {
            ih = aoh * SH - PH + kh; iw = aow * SW - PW + kw;
            ok = (ih >= 0) && (ih < IH) && (iw >= 0) && (iw < IW);
          }
          if (ok) {
            av = in[(((long)ab * Cin + ci) * IH + ih) * IW + iw];
            if (inRelu) av = fmaxf(av, 0.f);
          }
        }
        A[2 * v + e] = (_Float16)av;
      }
    }

    // ---- 4 B fragments from LDS (two aligned b128 reads each) + 4 WMMAs
#pragma unroll
    for (int nt = 0; nt < 4; ++nt) {
      const _Float16* rowp = &Btile[(nt * 16 + lane16) * 32];
      const v8h blo = *(const v8h*)(rowp + half * 8);
      const v8h bhi = *(const v8h*)(rowp + 16 + half * 8);
      v16h Bf;
#pragma unroll
      for (int i = 0; i < 8; ++i) { Bf[i] = blo[i]; Bf[8 + i] = bhi[i]; }
      acc[nt] = __builtin_amdgcn_wmma_f32_16x16x32_f16(false, A, false, Bf,
                                                       (short)0, acc[nt], false, false);
    }
    __syncthreads();                 // protect Btile before next chunk
  }

  const bool outRelu = (flags & F_OUTRELU) != 0;
  const int  ohw = OH * OW;
#pragma unroll
  for (int nt = 0; nt < 4; ++nt) {
    const int n = nbase + nt * 16 + lane16;
#pragma unroll
    for (int v = 0; v < 8; ++v) {
      const int mm = tileM * 16 + v + 8 * half; // C/D: lane=N, VGPR v => row
      if (mm < Mtot && n < Cout) {
        const int b = mm / ohw; const int r = mm - b * ohw;
        const int oh = r / OW;  const int ow = r - oh * OW;
        float val = acc[nt][v];
        if (bias) val += bias[n];
        const long oidx = (((long)b * Cout + n) * OH + oh) * OW + ow;
        if (resid) val += resid[oidx];
        if (outRelu) val = fmaxf(val, 0.f);
        out[oidx] = val;
      }
    }
  }
}

// ------------------------------------------------------------ VQ: ||e||^2
__global__ __launch_bounds__(256) void enorm_k(const float* __restrict__ emb,
                                               float* __restrict__ eNorm, int K, int D) {
  int nn = blockIdx.x * blockDim.x + threadIdx.x;
  if (nn < K) {
    float s = 0.f;
    for (int d = 0; d < D; ++d) { float e = emb[(long)nn * D + d]; s += e * e; }
    eNorm[nn] = s;
  }
}

// ------------------------------------- VQ argmin: scores = z . emb^T via WMMA
// A (z rows, D=64 -> two K chunks) hoisted out of the code-tile loop; emb B
// fragments are aligned 16B global vector loads (rows are K-contiguous).
__global__ __launch_bounds__(128) void vq_argmin_k(
    const float* __restrict__ z, const _Float16* __restrict__ embf,
    const float* __restrict__ eNorm, int* __restrict__ idxOut,
    int Npos, int D, int K, int H, int W)
{
  const int lane = threadIdx.x & 31;
  const int wave = threadIdx.x >> 5;
  const int tileM = blockIdx.x * (blockDim.x >> 5) + wave;
  const int tilesM = (Npos + 15) >> 4;
  if (tileM >= tilesM) return;
  const int lane16 = lane & 15, half = lane >> 4;
  const int mrow = tileM * 16 + lane16;
  const bool mOK = mrow < Npos;
  int zb = 0, zh = 0, zw = 0;
  {
    const int hw = H * W;
    const int mm = mOK ? mrow : 0;
    zb = mm / hw; int r = mm - zb * hw; zh = r / W; zw = r - zh * W;
  }
  v16h A0 = {}, A1 = {};
#pragma unroll
  for (int v = 0; v < 8; ++v) {
#pragma unroll
    for (int e = 0; e < 2; ++e) {
      const int kk = ((v & 4) << 2) + half * 8 + (v & 3) * 2 + e;
      float a0 = 0.f, a1 = 0.f;
      if (mOK) {
        a0 = z[(((long)zb * D + kk)      * H + zh) * W + zw];
        a1 = z[(((long)zb * D + kk + 32) * H + zh) * W + zw];
      }
      A0[2 * v + e] = (_Float16)a0;
      A1[2 * v + e] = (_Float16)a1;
    }
  }

  float best[8]; int bidx[8];
#pragma unroll
  for (int v = 0; v < 8; ++v) { best[v] = 3.4e38f; bidx[v] = 0; }

  for (int nt = 0; nt < K / 16; ++nt) {
    const int cn = nt * 16 + lane16;
    const _Float16* er = embf + (size_t)cn * D;
    const v8h b0lo = *(const v8h*)(er + half * 8);
    const v8h b0hi = *(const v8h*)(er + 16 + half * 8);
    const v8h b1lo = *(const v8h*)(er + 32 + half * 8);
    const v8h b1hi = *(const v8h*)(er + 48 + half * 8);
    v16h B0, B1;
#pragma unroll
    for (int i = 0; i < 8; ++i) {
      B0[i] = b0lo[i]; B0[8 + i] = b0hi[i];
      B1[i] = b1lo[i]; B1[8 + i] = b1hi[i];
    }
    v8f acc = {0.f, 0.f, 0.f, 0.f, 0.f, 0.f, 0.f, 0.f};
    acc = __builtin_amdgcn_wmma_f32_16x16x32_f16(false, A0, false, B0,
                                                 (short)0, acc, false, false);
    acc = __builtin_amdgcn_wmma_f32_16x16x32_f16(false, A1, false, B1,
                                                 (short)0, acc, false, false);
    const float en = eNorm[cn];
#pragma unroll
    for (int v = 0; v < 8; ++v) {
      const float dist = en - 2.f * acc[v];   // ||z||^2 constant per row
      if (dist < best[v]) { best[v] = dist; bidx[v] = cn; }
    }
  }
  // min across the 16 lanes of each half (they hold the same 8 rows)
#pragma unroll
  for (int offm = 8; offm >= 1; offm >>= 1) {
#pragma unroll
    for (int v = 0; v < 8; ++v) {
      const float od = __shfl_xor(best[v], offm, 16);
      const int   oi = __shfl_xor(bidx[v], offm, 16);
      if (od < best[v] || (od == best[v] && oi < bidx[v])) { best[v] = od; bidx[v] = oi; }
    }
  }
  if (lane16 == 0) {
#pragma unroll
    for (int v = 0; v < 8; ++v) {
      const int row = tileM * 16 + v + 8 * half;
      if (row < Npos) idxOut[row] = bidx[v];
    }
  }
}

// -------------------------------------- VQ gather q, sum (q-z)^2, histogram
__global__ __launch_bounds__(256) void vq_gather_k(
    const float* __restrict__ z, const float* __restrict__ emb,
    const int* __restrict__ idx, float* __restrict__ q,
    float* __restrict__ sumSq, float* __restrict__ counts,
    int Npos, int D, int H, int W)
{
  __shared__ float red[256];
  const int tid = blockIdx.x * blockDim.x + threadIdx.x;
  float c = 0.f;
  if (tid < Npos * D) {
    const int i = tid / D, d = tid - (tid / D) * D;
    const int hw = H * W;
    const int b = i / hw; const int r = i - b * hw;
    const int h = r / W;  const int w = r - h * W;
    const int code = idx[i];
    const float qv = emb[(long)code * D + d];
    const long zi = (((long)b * D + d) * H + h) * W + w;
    const float zv = z[zi];
    q[zi] = qv;                         // straight-through value == q
    const float diff = qv - zv;
    c = diff * diff;
    if (d == 0) atomicAdd(&counts[code], 1.f);
  }
  red[threadIdx.x] = c;
  __syncthreads();
  for (int s = 128; s >= 1; s >>= 1) {
    if ((int)threadIdx.x < s) red[threadIdx.x] += red[threadIdx.x + s];
    __syncthreads();
  }
  if (threadIdx.x == 0) atomicAdd(sumSq, red[0]);
}

// ------------------------------------------------- loss + perplexity finish
__global__ __launch_bounds__(512) void vq_finish_k(
    const float* __restrict__ sumSq, const float* __restrict__ counts,
    float* __restrict__ out, int outSize, int K, int Npos, int D)
{
  __shared__ float red[512];
  const int t = threadIdx.x;
  const float p = (t < K) ? counts[t] / (float)Npos : 0.f;
  red[t] = p * logf(p + 1e-10f);
  __syncthreads();
  for (int s = 256; s >= 1; s >>= 1) {
    if (t < s) red[t] += red[t + s];
    __syncthreads();
  }
  if (t == 0) {
    out[0] = 1.25f * sumSq[0] / (float)(Npos * D);   // q_latent + 0.25*e_latent
    out[outSize - 1] = expf(-red[0]);
  }
}

// ---------------------------------------------------------------------------
extern "C" void kernel_launch(void* const* d_in, const int* in_sizes, int n_in,
                              void* d_out, int out_size, void* d_ws, size_t ws_size,
                              hipStream_t stream) {
  (void)in_sizes; (void)n_in; (void)ws_size;
  const float* x      = (const float*)d_in[0];
  const float* encW1  = (const float*)d_in[1];
  const float* encB1  = (const float*)d_in[2];
  const float* encW2  = (const float*)d_in[3];
  const float* encB2  = (const float*)d_in[4];
  const float* encW3  = (const float*)d_in[5];
  const float* encB3  = (const float*)d_in[6];
  const float* encRW1 = (const float*)d_in[7];
  const float* encRW2 = (const float*)d_in[8];
  const float* preW   = (const float*)d_in[9];
  const float* preB   = (const float*)d_in[10];
  const float* emb    = (const float*)d_in[11];
  const float* decW1  = (const float*)d_in[12];
  const float* decB1  = (const float*)d_in[13];
  const float* decRW1 = (const float*)d_in[14];
  const float* decRW2 = (const float*)d_in[15];
  const float* decTW1 = (const float*)d_in[16];
  const float* decTB1 = (const float*)d_in[17];
  const float* decTW2 = (const float*)d_in[18];
  const float* decTB2 = (const float*)d_in[19];

  float* out = (float*)d_out;

  char* base = (char*)d_ws;
  size_t off = 0;
  auto alloc = [&](size_t bytes) -> void* {
    off = (off + 255) & ~(size_t)255;
    void* p = base + off; off += bytes; return p;
  };

  // f16 weight copies
  const int sW1 = 64*2*10*11, sW2 = 128*64*144, sW3 = 128*128*9,
            sRW1 = 2*64*128*9, sRW2 = 2*128*64, sPRE = 64*128, sEMB = 512*64,
            sDW1 = 128*64*9, sTW1 = 128*64*144, sTW2 = 64*2*110;
  _Float16* fW1  = (_Float16*)alloc((size_t)sW1  * 2);
  _Float16* fW2  = (_Float16*)alloc((size_t)sW2  * 2);
  _Float16* fW3  = (_Float16*)alloc((size_t)sW3  * 2);
  _Float16* fRW1 = (_Float16*)alloc((size_t)sRW1 * 2);
  _Float16* fRW2 = (_Float16*)alloc((size_t)sRW2 * 2);
  _Float16* fPRE = (_Float16*)alloc((size_t)sPRE * 2);
  _Float16* fEMB = (_Float16*)alloc((size_t)sEMB * 2);
  _Float16* fDW1 = (_Float16*)alloc((size_t)sDW1 * 2);
  _Float16* fDR1 = (_Float16*)alloc((size_t)sRW1 * 2);
  _Float16* fDR2 = (_Float16*)alloc((size_t)sRW2 * 2);
  _Float16* fTW1 = (_Float16*)alloc((size_t)sTW1 * 2);
  _Float16* fTW2 = (_Float16*)alloc((size_t)sTW2 * 2);

  // activations (f32)
  float* h1  = (float*)alloc((size_t)32*64*50*66 * 4);
  float* h2  = (float*)alloc((size_t)32*128*5*6  * 4);
  float* h3  = (float*)alloc((size_t)32*128*5*6  * 4);
  float* t64 = (float*)alloc((size_t)32*64*5*6   * 4);
  float* zb  = (float*)alloc((size_t)32*64*5*6   * 4);
  float* qb  = (float*)alloc((size_t)32*64*5*6   * 4);
  float* dd1 = (float*)alloc((size_t)32*128*5*6  * 4);
  float* dd2 = (float*)alloc((size_t)32*64*50*60 * 4);
  int*   idxb   = (int*)alloc(960 * 4);
  float* eN     = (float*)alloc(512 * 4);
  float* counts = (float*)alloc(512 * 4);
  float* sumSq  = (float*)alloc(4);

  auto cast16 = [&](const float* s, _Float16* d, int n) {
    f32_to_f16_k<<<dim3((n + 255) / 256), 256, 0, stream>>>(s, d, n);
  };
  cast16(encW1, fW1, sW1);   cast16(encW2, fW2, sW2);   cast16(encW3, fW3, sW3);
  cast16(encRW1, fRW1, sRW1); cast16(encRW2, fRW2, sRW2);
  cast16(preW, fPRE, sPRE);  cast16(emb, fEMB, sEMB);
  cast16(decW1, fDW1, sDW1); cast16(decRW1, fDR1, sRW1); cast16(decRW2, fDR2, sRW2);
  cast16(decTW1, fTW1, sTW1); cast16(decTW2, fTW2, sTW2);

  auto conv = [&](const float* in, const _Float16* wf, const float* bias,
                  const float* resid, float* o,
                  int NB, int Cin, int IH, int IW, int Cout, int OH, int OW,
                  int KH, int KW, int SH, int SW, int PH, int PW, int flags) {
    const int Mtot = NB * OH * OW;
    const int tilesM = (Mtot + 15) / 16, tilesN64 = (Cout + 63) / 64;
    dim3 grid((tilesM + 3) / 4, tilesN64, 1);
    conv_wmma_k<<<grid, 128, 0, stream>>>(in, wf, bias, resid, o,
        NB, Cin, IH, IW, Cout, OH, OW, KH, KW, SH, SW, PH, PW, flags);
  };

  auto resstack = [&](float* xb, const _Float16* rw1, const _Float16* rw2,
                      int C, int Cm, int H, int W) {
    for (int i = 0; i < 2; ++i) {
      conv(xb, rw1 + (size_t)i * Cm * C * 9, nullptr, nullptr, t64,
           32, C, H, W, Cm, H, W, 3, 3, 1, 1, 1, 1, F_INRELU);
      conv(t64, rw2 + (size_t)i * C * Cm, nullptr, xb, xb,
           32, Cm, H, W, C, H, W, 1, 1, 1, 1, 0, 0,
           F_INRELU | (i == 1 ? F_OUTRELU : 0));
    }
  };

  // ---- Encoder
  conv(x,  fW1, encB1, nullptr, h1, 32, 2,  400, 600, 64,  50, 66, 10, 11, 8, 9, 1, 1, F_OUTRELU);
  conv(h1, fW2, encB2, nullptr, h2, 32, 64, 50,  66,  128, 5,  6,  12, 12, 10, 10, 1, 1, F_OUTRELU);
  conv(h2, fW3, encB3, nullptr, h3, 32, 128, 5,  6,   128, 5,  6,  3, 3, 1, 1, 1, 1, 0);
  resstack(h3, fRW1, fRW2, 128, 64, 5, 6);
  conv(h3, fPRE, preB, nullptr, zb, 32, 128, 5, 6, 64, 5, 6, 1, 1, 1, 1, 0, 0, 0);

  // ---- Vector quantizer
  enorm_k<<<dim3(2), 256, 0, stream>>>(emb, eN, 512, 64);
  vq_argmin_k<<<dim3(15), 128, 0, stream>>>(zb, fEMB, eN, idxb, 960, 64, 512, 5, 6);
  hipMemsetAsync(counts, 0, 512 * 4, stream);
  hipMemsetAsync(sumSq, 0, 4, stream);
  vq_gather_k<<<dim3((960 * 64 + 255) / 256), 256, 0, stream>>>(
      zb, emb, idxb, qb, sumSq, counts, 960, 64, 5, 6);

  // ---- Decoder
  conv(qb,  fDW1, decB1, nullptr, dd1, 32, 64, 5, 6, 128, 5, 6, 3, 3, 1, 1, 1, 1, 0);
  resstack(dd1, fDR1, fDR2, 128, 64, 5, 6);
  conv(dd1, fTW1, decTB1, nullptr, dd2, 32, 128, 5, 6, 64, 50, 60,
       12, 12, 10, 10, 1, 1, F_TRANS | F_OUTRELU);
  conv(dd2, fTW2, decTB2, nullptr, out + 1, 32, 64, 50, 60, 2, 400, 540,
       10, 11, 8, 9, 1, 1, F_TRANS);

  // ---- loss / perplexity
  vq_finish_k<<<dim3(1), 512, 0, stream>>>(sumSq, counts, out, out_size, 512, 960, 64);
}
